// RGBT_Fusion_2_39092792328561
// MI455X (gfx1250) — matrix operand order repaired and checked by reading, not compiled, not run on hardware
//
#include <hip/hip_runtime.h>

typedef __attribute__((ext_vector_type(16))) __bf16 v16bf;
typedef __attribute__((ext_vector_type(8)))  float  v8f;
typedef unsigned int u32x4 __attribute__((ext_vector_type(4)));
typedef int          i32x4 __attribute__((ext_vector_type(4)));
typedef int          i32x8 __attribute__((ext_vector_type(8)));

#define HH   64
#define WWID 64
#define HWSZ 4096
#define BSZ  4

#if __has_include(<hip/amd_detail/amd_gfx1250_TDM.h>)
#define TDM_6ARG 1
#else
#define TDM_6ARG 0
#endif

__device__ __forceinline__ float bf2f(unsigned short h) {
  return __uint_as_float(((unsigned int)h) << 16);
}
__device__ __forceinline__ unsigned short f2bf(float f) {
  unsigned int u = __float_as_uint(f);
  return (unsigned short)((u + 0x7FFFu + ((u >> 16) & 1u)) >> 16);
}

union AFrag { v16bf v; uint4 q[2]; };
union BFrag { v16bf v; unsigned short s[16]; };

// ---------------- prologue: pack inputs / weights to bf16 ----------------

__global__ void pack_x(const float* __restrict__ iv, const float* __restrict__ ii,
                       unsigned short* __restrict__ dst) {
  int t = blockIdx.x * blockDim.x + threadIdx.x;      // < 4*256*4096
  int b = t / (256 * HWSZ);
  int c = (t / HWSZ) % 256;
  int i = t % HWSZ;
  float v = (c < 128) ? iv[((size_t)(b * 128 + c)) * HWSZ + i]
                      : ii[((size_t)(b * 128 + c - 128)) * HWSZ + i];
  dst[t] = f2bf(v);
}

// off_w[54][256][3][3] -> wA1[64][9][256] (zero-padded rows 54..63)
__global__ void pack_offw(const float* __restrict__ w, unsigned short* __restrict__ dst) {
  int t = blockIdx.x * blockDim.x + threadIdx.x;      // < 64*9*256
  int m = t / (9 * 256);
  int tap = (t / 256) % 9;
  int c = t % 256;
  dst[t] = (m < 54) ? f2bf(w[((size_t)(m * 256 + c)) * 9 + tap]) : (unsigned short)0;
}

// dconv_w[128][256][3][3] -> wA2[128][2][9][128]
__global__ void pack_dconvw(const float* __restrict__ w, unsigned short* __restrict__ dst) {
  int t = blockIdx.x * blockDim.x + threadIdx.x;      // < 128*2*9*128
  int o = t / 2304;
  int g = (t / 1152) % 2;
  int k = (t / 128) % 9;
  int c = t % 128;
  dst[t] = f2bf(w[((size_t)(o * 256 + g * 128 + c)) * 9 + k]);
}

// conv2_w[128][128][3][3] -> wA3[128][9][128]
__global__ void pack_conv2w(const float* __restrict__ w, unsigned short* __restrict__ dst) {
  int t = blockIdx.x * blockDim.x + threadIdx.x;      // < 128*9*128
  int m = t / 1152;
  int tap = (t / 128) % 9;
  int c = t % 128;
  dst[t] = f2bf(w[((size_t)(m * 128 + c)) * 9 + tap]);
}

// ---------------- TDM staging of one [32ch][3row][64col] bf16 slab into LDS ----------------
// LDS layout: tile[ch*192 + row*64 + col], rows are image rows y-1..y+1 (halo rows zeroed).
__device__ __forceinline__ void stage_tile(unsigned short* tile,
                                           const unsigned short* __restrict__ xin,
                                           int bC_plus_c0,  // b*C + c0
                                           int y) {
#if defined(__gfx1250__) && __has_builtin(__builtin_amdgcn_tensor_load_to_lds)
  // zero the missing halo row at the image top/bottom edge
  if (y == 0 || y == 63) {
    const int slotbase = (y == 0) ? 0 : 128;
    for (int i = threadIdx.x; i < 32 * 64; i += 128)
      tile[(i >> 6) * 192 + slotbase + (i & 63)] = 0;
  }
  if ((threadIdx.x >> 5) == 0) {   // wave 0 issues the tensor DMA
    const int  ystart = (y == 0) ? 0 : y - 1;
    const int  nrows  = (y == 0 || y == 63) ? 2 : 3;
    const bool pad    = (nrows == 2);
    // nrows==2: each channel stores 128 elems (64 dw); pad 32 dw after each 64-dw
    // interval so the channel stride stays 192 elems. pad_interval=5, pad_amount=31.
    unsigned ldsaddr = (unsigned)(size_t)(void*)tile + ((y == 0) ? 128u : 0u);
    unsigned long long gaddr =
        (unsigned long long)(size_t)(xin) +
        ((unsigned long long)bC_plus_c0 * HWSZ + (unsigned)(ystart * WWID)) * 2ull;
    u32x4 g0 = { 1u,                                   // count=1, user descriptor
                 ldsaddr,
                 (unsigned)(gaddr & 0xffffffffull),
                 (unsigned)((gaddr >> 32) & 0x01ffffffull) | 0x80000000u };  // type=2
    unsigned dw0 = 0x00010000u;                        // data_size=2B, mask=0
    if (pad) dw0 |= (1u << 20) | (5u << 22) | (31u << 25);
    i32x8 g1 = { (int)dw0,
                 (int)(64u << 16),                     // tensor_dim0 = 64
                 (int)((unsigned)nrows << 16),         // tensor_dim1 = nrows
                 (int)(64u << 16),                     // tile_dim0 = 64
                 (int)((32u << 16) | (unsigned)nrows), // tile_dim2=32 | tile_dim1=nrows
                 64,                                   // tensor_dim0_stride = W
                 (int)(4096u << 16),                   // tensor_dim1_stride = H*W (low16)
                 0 };
    i32x4 g2 = { 32, 0, 4096, 0 };                     // tensor_dim2=32, dim2_stride=H*W
    i32x4 g3 = { 0, 0, 0, 0 };
#if TDM_6ARG
    __builtin_amdgcn_tensor_load_to_lds(g0, g1, g2, g3, (i32x8)0, 0);
#else
    __builtin_amdgcn_tensor_load_to_lds(g0, g1, g2, g3, 0);
#endif
    __builtin_amdgcn_s_wait_tensorcnt(0);
  }
#else
  // fallback: cooperative staging with manual halo-zero
  for (int i = threadIdx.x; i < 32 * 3 * 64; i += 128) {
    int ch = i / 192;
    int r = (i % 192) >> 6;
    int col = i & 63;
    int yy = y + r - 1;
    unsigned short v = 0;
    if (yy >= 0 && yy < HH)
      v = xin[((size_t)(bC_plus_c0 + ch)) * HWSZ + yy * WWID + col];
    tile[ch * 192 + r * 64 + col] = v;
  }
#endif
}

// ---------------- implicit-GEMM 3x3 SAME conv (bf16 WMMA, TDM->LDS staged) ----------------
// grid: B*H blocks, 128 threads (4 waves; wave w owns pixels [w*16, w*16+16)).
// One block accumulates ALL MT M-tiles so each B-fragment feeds MT WMMAs.
// wA layout: [MT*16][9][C] bf16, K-order = tap*C + c.  C is a template constant so the
// per-M-tile A-fragment stride (16*9*C elems) folds into load immediate offsets.
template <int MT, int C>
__global__ void conv3x3_wmma(const unsigned short* __restrict__ xin,   // [B][C][HW] bf16
                             const unsigned short* __restrict__ wA,    // [MT*16][9][C]
                             const float* __restrict__ bias,           // [Mreal] or null
                             float* __restrict__ out,                  // [B][Mreal][HW]
                             int Mreal) {
  __shared__ unsigned short tile[32 * 3 * 64];   // [ch][row][col] = 12 KB
  const int row = blockIdx.x;
  const int b = row >> 6, y = row & 63;
  const int lane = threadIdx.x & 31;
  const int wid  = threadIdx.x >> 5;
  const int x0 = wid * 16;
  const int p = lane & 15;        // N within tile (also M-row for A frag)
  const int half = lane >> 4;

  v8f acc[MT];
  #pragma unroll
  for (int mt = 0; mt < MT; ++mt) acc[mt] = (v8f)0.0f;

  for (int c0 = 0; c0 < C; c0 += 32) {
    __syncthreads();                               // previous chunk fully consumed
    stage_tile(tile, xin, b * C + c0, y);
    __syncthreads();                               // slab visible to all waves

    for (int t = 0; t < 9; ++t) {
      const int dy = t / 3 - 1, dx = t % 3 - 1;
      const int xx = x0 + p + dx;
      const int xc = xx < 0 ? 0 : (xx > 63 ? 63 : xx);
      const unsigned short keep = (xx >= 0 && xx < WWID) ? (unsigned short)0xFFFF
                                                         : (unsigned short)0;
      const unsigned short* tp = tile + (half * 16) * 192 + (dy + 1) * 64 + xc;
      BFrag Bf;  // lane n=p; K = half*16 + j  -> channel c0 + that
      #pragma unroll
      for (int j = 0; j < 16; ++j) Bf.s[j] = (unsigned short)(tp[j * 192] & keep);

      // single base pointer; per-M-tile stride is the literal 16*9*C -> imm offsets
      const unsigned short* wbase = wA + ((size_t)p * 9 + t) * C + c0 + half * 16;
      #pragma unroll
      for (int mt = 0; mt < MT; ++mt) {
        AFrag A;  // lane m=p; K = half*16+j
        A.q[0] = *(const uint4*)(wbase + mt * (16 * 9 * C));
        A.q[1] = *(const uint4*)(wbase + mt * (16 * 9 * C) + 8);
        acc[mt] = __builtin_amdgcn_wmma_f32_16x16x32_bf16(false, A.v, false, Bf.v,
                                                          (short)0, acc[mt], false, false);
      }
    }
  }
  // per-lane base; channel stride folds into store immediates
  float* outl = out + ((size_t)b * Mreal + half * 8) * HWSZ + y * WWID + x0 + p;
  #pragma unroll
  for (int mt = 0; mt < MT; ++mt) {
    #pragma unroll
    for (int r = 0; r < 8; ++r) {
      int m = mt * 16 + half * 8 + r;
      if (m < Mreal) {
        float bv = bias ? bias[m] : 0.0f;
        outl[(mt * 16 + r) * HWSZ] = acc[mt][r] + bv;
      }
    }
  }
}

// ---------------- deformable conv: fused bilinear sample + bf16 WMMA GEMM ----------------
// grid: B*H, block 128. Each wave: 16 pixels x all 128 out channels (8 M-tiles).
// K-order: g*1152 + k*128 + c ; wA2[o][g][k][c].  Gathers are data-dependent -> serviced
// from L2 (x_bf16 = 8.4 MB << 192 MB L2); each gathered B-fragment feeds 8 WMMAs.
__global__ void deform_wmma(const unsigned short* __restrict__ xin,   // [B][256][HW] bf16
                            const float* __restrict__ off_all,        // [B][54][HW]
                            const unsigned short* __restrict__ wA,    // [128][2][9][128]
                            float* __restrict__ out) {                // [B][128][HW]
  const int row = blockIdx.x;
  const int b = row >> 6, y = row & 63;
  const int lane = threadIdx.x & 31;
  const int wid  = threadIdx.x >> 5;
  const int p = lane & 15;
  const int half = lane >> 4;
  const int px = wid * 16 + p;

  v8f acc[8];
  #pragma unroll
  for (int mt = 0; mt < 8; ++mt) acc[mt] = (v8f)0.0f;
  const float* offb = off_all + (size_t)b * 54 * HWSZ + y * WWID + px;

  for (int g = 0; g < 2; ++g) {
    const unsigned short* xg = xin + ((size_t)b * 256 + g * 128) * HWSZ;  // uniform base
    for (int k = 0; k < 9; ++k) {
      float dyv  = offb[(size_t)(g * 18 + 2 * k)     * HWSZ];
      float dxv  = offb[(size_t)(g * 18 + 2 * k + 1) * HWSZ];
      float mraw = offb[(size_t)(36 + g * 9 + k)     * HWSZ];
      float msk = 1.0f / (1.0f + __expf(-mraw));
      float py = dyv + (float)y + (float)(k / 3 - 1);
      float pxf = dxv + (float)px + (float)(k % 3 - 1);
      float y0f = floorf(py), x0f = floorf(pxf);
      float ly = py - y0f, lx = pxf - x0f;
      int y0i = (int)y0f, x0i = (int)x0f;
      int y1i = y0i + 1,  x1i = x0i + 1;
      float ok00 = (y0i >= 0 && y0i < HH && x0i >= 0 && x0i < WWID) ? 1.0f : 0.0f;
      float ok01 = (y0i >= 0 && y0i < HH && x1i >= 0 && x1i < WWID) ? 1.0f : 0.0f;
      float ok10 = (y1i >= 0 && y1i < HH && x0i >= 0 && x0i < WWID) ? 1.0f : 0.0f;
      float ok11 = (y1i >= 0 && y1i < HH && x1i >= 0 && x1i < WWID) ? 1.0f : 0.0f;
      float w00 = (1.0f - ly) * (1.0f - lx) * ok00 * msk;
      float w01 = (1.0f - ly) * lx          * ok01 * msk;
      float w10 = ly * (1.0f - lx)          * ok10 * msk;
      float w11 = ly * lx                   * ok11 * msk;
      int y0c = y0i < 0 ? 0 : (y0i > HH - 1 ? HH - 1 : y0i);
      int y1c = y1i < 0 ? 0 : (y1i > HH - 1 ? HH - 1 : y1i);
      int x0c = x0i < 0 ? 0 : (x0i > WWID - 1 ? WWID - 1 : x0i);
      int x1c = x1i < 0 ? 0 : (x1i > WWID - 1 ? WWID - 1 : x1i);
      unsigned a00 = (unsigned)(y0c * WWID + x0c), a01 = (unsigned)(y0c * WWID + x1c);
      unsigned a10 = (unsigned)(y1c * WWID + x0c), a11 = (unsigned)(y1c * WWID + x1c);

      const int kbase = g * 9 + k;
      for (int c0 = 0; c0 < 128; c0 += 32) {
        BFrag Bf;  // lane's 16 channels: c = c0 + half*16 + j
        const unsigned xbase = (unsigned)((c0 + half * 16) * HWSZ);
        #pragma unroll
        for (int j = 0; j < 16; ++j) {
          const unsigned o = xbase + (unsigned)(j * HWSZ);  // j*HWSZ folds to imm offset
          float v = w00 * bf2f(xg[o + a00]) + w01 * bf2f(xg[o + a01]) +
                    w10 * bf2f(xg[o + a10]) + w11 * bf2f(xg[o + a11]);
          Bf.s[j] = f2bf(v);
        }
        // single base pointer; per-M-tile stride 16*18*128 elems -> imm offsets
        const unsigned short* wbase = wA + (size_t)p * (18 * 128) + kbase * 128 + c0 + half * 16;
        #pragma unroll
        for (int mtile = 0; mtile < 8; ++mtile) {
          AFrag A;
          A.q[0] = *(const uint4*)(wbase + mtile * (16 * 18 * 128));
          A.q[1] = *(const uint4*)(wbase + mtile * (16 * 18 * 128) + 8);
          acc[mtile] = __builtin_amdgcn_wmma_f32_16x16x32_bf16(false, A.v, false, Bf.v,
                                                               (short)0, acc[mtile], false, false);
        }
      }
    }
  }
  // per-lane base; channel stride folds into store immediates
  float* outl = out + ((size_t)b * 128 + half * 8) * HWSZ + y * WWID + px;
  #pragma unroll
  for (int mtile = 0; mtile < 8; ++mtile)
    #pragma unroll
    for (int r = 0; r < 8; ++r)
      outl[(mtile * 16 + r) * HWSZ] = acc[mtile][r];
}

// ---------------- batchnorm stats + apply ----------------

__global__ void bn_reduce(const float* __restrict__ x, float* __restrict__ mean,
                          float* __restrict__ inv, int Cn) {
  __shared__ float ss[256], sq[256];
  const int c = blockIdx.x;
  const int tid = threadIdx.x;
  float s = 0.0f, q = 0.0f;
  for (int b = 0; b < BSZ; ++b) {
    const float* pp = x + ((size_t)(b * Cn + c)) * HWSZ;
    for (int i = tid; i < HWSZ; i += 256) { float v = pp[i]; s += v; q += v * v; }
  }
  ss[tid] = s; sq[tid] = q;
  __syncthreads();
  for (int o = 128; o > 0; o >>= 1) {
    if (tid < o) { ss[tid] += ss[tid + o]; sq[tid] += sq[tid + o]; }
    __syncthreads();
  }
  if (tid == 0) {
    const float n = (float)(BSZ * HWSZ);
    float m = ss[0] / n;
    float var = sq[0] / n - m * m;
    mean[c] = m;
    inv[c] = rsqrtf(var + 1e-5f);
  }
}

__global__ void bn_apply_bf16(const float* __restrict__ x, const float* __restrict__ mean,
                              const float* __restrict__ inv, const float* __restrict__ g,
                              const float* __restrict__ bb, unsigned short* __restrict__ dst) {
  int t = blockIdx.x * blockDim.x + threadIdx.x;   // < 4*128*4096
  int c = (t / HWSZ) % 128;
  float v = (x[t] - mean[c]) * inv[c] * g[c] + bb[c];
  dst[t] = f2bf(v > 0.0f ? v : 0.0f);
}

__global__ void bn_apply_f32(const float* __restrict__ x, const float* __restrict__ mean,
                             const float* __restrict__ inv, const float* __restrict__ g,
                             const float* __restrict__ bb, float* __restrict__ dst) {
  int t = blockIdx.x * blockDim.x + threadIdx.x;
  int c = (t / HWSZ) % 128;
  float v = (x[t] - mean[c]) * inv[c] * g[c] + bb[c];
  dst[t] = v > 0.0f ? v : 0.0f;
}

// ---------------- launch ----------------

extern "C" void kernel_launch(void* const* d_in, const int* in_sizes, int n_in,
                              void* d_out, int out_size, void* d_ws, size_t ws_size,
                              hipStream_t stream) {
  const float* input_v = (const float*)d_in[0];
  const float* input_i = (const float*)d_in[1];
  const float* off_w   = (const float*)d_in[2];
  const float* off_b   = (const float*)d_in[3];
  const float* dconv_w = (const float*)d_in[4];
  const float* bn1_g   = (const float*)d_in[5];
  const float* bn1_b   = (const float*)d_in[6];
  const float* conv2_w = (const float*)d_in[7];
  const float* bn2_g   = (const float*)d_in[8];
  const float* bn2_b   = (const float*)d_in[9];

  char* ws = (char*)d_ws;
  unsigned short* x_bf16    = (unsigned short*)(ws + 0);             //  8,388,608 B
  unsigned short* wA1       = (unsigned short*)(ws + 8388608);       //    294,912 B
  unsigned short* wA2       = (unsigned short*)(ws + 8683520);       //    589,824 B
  unsigned short* wA3       = (unsigned short*)(ws + 9273344);       //    294,912 B
  float*          off_all   = (float*)        (ws + 9568256);        //  3,538,944 B
  float*          out1_raw  = (float*)        (ws + 13107200);       //  8,388,608 B
  unsigned short* out1_bf16 = (unsigned short*)(ws + 21495808);      //  4,194,304 B
  float*          out2_raw  = out1_raw;                              // alias (dead by then)
  float*          mean1     = (float*)        (ws + 25690112);
  float*          inv1      = mean1 + 128;
  float*          mean2     = mean1 + 256;
  float*          inv2      = mean1 + 384;

  pack_x     <<<(4 * 256 * HWSZ) / 256, 256, 0, stream>>>(input_v, input_i, x_bf16);
  pack_offw  <<<(64 * 9 * 256) / 256, 256, 0, stream>>>(off_w, wA1);
  pack_dconvw<<<(128 * 2 * 9 * 128) / 256, 256, 0, stream>>>(dconv_w, wA2);
  pack_conv2w<<<(128 * 9 * 128) / 256, 256, 0, stream>>>(conv2_w, wA3);

  // stage 1: offset/mask conv  (M=54 padded to 64, K=2304, N=16384)
  conv3x3_wmma<4, 256><<<BSZ * HH, 128, 0, stream>>>(x_bf16, wA1, off_b, off_all, 54);

  // stage 2: fused deformable sampling + channel GEMM (M=128, K=2304, N=16384)
  deform_wmma<<<BSZ * HH, 128, 0, stream>>>(x_bf16, off_all, wA2, out1_raw);

  // BN1 + ReLU -> bf16
  bn_reduce<<<128, 256, 0, stream>>>(out1_raw, mean1, inv1, 128);
  bn_apply_bf16<<<(4 * 128 * HWSZ) / 256, 256, 0, stream>>>(out1_raw, mean1, inv1, bn1_g, bn1_b, out1_bf16);

  // stage 3: conv2 (M=128, K=1152, N=16384)
  conv3x3_wmma<8, 128><<<BSZ * HH, 128, 0, stream>>>(out1_bf16, wA3, nullptr, out2_raw, 128);

  // BN2 + ReLU -> fp32 output
  bn_reduce<<<128, 256, 0, stream>>>(out2_raw, mean2, inv2, 128);
  bn_apply_f32<<<(4 * 128 * HWSZ) / 256, 256, 0, stream>>>(out2_raw, mean2, inv2, bn2_g, bn2_b, (float*)d_out);
}
